// MANN_LSTMCell_76020921140091
// MI455X (gfx1250) — matrix operand order repaired and checked by reading, not compiled, not run on hardware
//
#include <hip/hip_runtime.h>
#include <hip/hip_bf16.h>
#include <math.h>

// Problem constants (reference: M, B, U, D)
static constexpr int MM = 65536;
static constexpr int BB = 128;
static constexpr int UU = 256;
static constexpr int DD = 512;

typedef __attribute__((ext_vector_type(16))) _Float16 v16h;
typedef __attribute__((ext_vector_type(8)))  float    v8f;

union Vh { v16h v; _Float16 h[16]; };
union Vf { v8f  v; float     f[8];  };

__device__ __forceinline__ v8f wmma32f16(v16h a, v16h b, v8f c) {
  // D = A(16x32 f16) * B(32x16 f16) + C(16x16 f32)
  return __builtin_amdgcn_wmma_f32_16x16x32_f16(
      /*neg_a=*/false, a, /*neg_b=*/false, b,
      /*c_mod=*/(short)0, c, /*reuse_a=*/false, /*reuse_b=*/false);
}

// 16-bit A-matrix 16x32 layout (ISA 7.12.2): lane = row (M), element e -> K index
__device__ __forceinline__ int kmapA(int e, int hi) {
  int v = e >> 1, c = e & 1;
  return (v & 3) * 2 + c + (hi ? 8 : 0) + ((v >= 4) ? 16 : 0);
}
// Inverse of kmapA: k (0..31) -> {element index e, lane half hi}
__device__ __forceinline__ void invA(int kk, int& e, int& hi) {
  hi = (kk >> 3) & 1;
  e  = (kk & 7) | ((kk >> 4) << 3);
}
// 16-bit B-matrix 32x16 layout: lane = col (N), same K order as A (hi half -> +8, e>=8 -> +16)
__device__ __forceinline__ int kmapB(int e, int hi) {
  int v = e >> 1, c = e & 1;
  return (v & 3) * 2 + c + (hi ? 8 : 0) + ((v >= 4) ? 16 : 0);
}
// Inverse of kmapB (same as invA)
__device__ __forceinline__ void invB(int kk, int& e, int& hi) {
  hi = (kk >> 3) & 1;
  e  = (kk & 7) | ((kk >> 4) << 3);
}

__device__ __forceinline__ float sigmoidf_(float x) { return 1.f / (1.f + expf(-x)); }

// Fragment-order addressing: fragment f, lane l -> 16 contiguous halves (32 B aligned)
__device__ __forceinline__ int fragIdx(int f, int lane) { return (f * 32 + lane) * 16; }

// ---------------------------------------------------------------------------
// K1: z = [inputs | r_tm1 | h_tm1] @ [kernel ; rec_kernel]   ([128 x 1024])
// One wave per 16x16 output tile; k-loop split at source boundaries (512, 768)
// so all pointers are loop-invariant (no per-element branching).
// ---------------------------------------------------------------------------
__global__ void __launch_bounds__(128) k_lstm_gemm(
    const float* __restrict__ inputs, const float* __restrict__ r_tm1,
    const float* __restrict__ h_tm1, const float* __restrict__ kernelW,
    const float* __restrict__ recW, float* __restrict__ z) {
  int lane = threadIdx.x & 31, wave = threadIdx.x >> 5, hi = lane >> 4;
  int tile = blockIdx.x * 4 + wave;            // 512 tiles total
  int btile = tile >> 6, ntile = tile & 63;    // 8 x 64
  int row  = btile * 16 + (lane & 15);
  int ncol = ntile * 16 + (lane & 15);

  Vf acc;
#pragma unroll
  for (int j = 0; j < 8; ++j) acc.f[j] = 0.f;

  // ks 0..15: A <- inputs (cols 0..511), B <- kernel rows 0..511
  for (int ks = 0; ks < 16; ++ks) {
    Vh a, b;
#pragma unroll
    for (int e = 0; e < 16; ++e) {
      a.h[e] = (_Float16)inputs[row * 512 + ks * 32 + kmapA(e, hi)];
      b.h[e] = (_Float16)kernelW[(ks * 32 + kmapB(e, hi)) * 1024 + ncol];
    }
    acc.v = wmma32f16(a.v, b.v, acc.v);
  }
  // ks 16..23: A <- r_tm1 (cols 512..767), B <- kernel rows 512..767
  for (int ks = 0; ks < 8; ++ks) {
    Vh a, b;
#pragma unroll
    for (int e = 0; e < 16; ++e) {
      a.h[e] = (_Float16)r_tm1[row * 256 + ks * 32 + kmapA(e, hi)];
      b.h[e] = (_Float16)kernelW[(512 + ks * 32 + kmapB(e, hi)) * 1024 + ncol];
    }
    acc.v = wmma32f16(a.v, b.v, acc.v);
  }
  // ks 24..31: A <- h_tm1 (cols 768..1023), B <- rec_kernel rows 0..255
  for (int ks = 0; ks < 8; ++ks) {
    Vh a, b;
#pragma unroll
    for (int e = 0; e < 16; ++e) {
      a.h[e] = (_Float16)h_tm1[row * 256 + ks * 32 + kmapA(e, hi)];
      b.h[e] = (_Float16)recW[(ks * 32 + kmapB(e, hi)) * 1024 + ncol];
    }
    acc.v = wmma32f16(a.v, b.v, acc.v);
  }
#pragma unroll
  for (int v = 0; v < 8; ++v) {
    int m = btile * 16 + v + 8 * hi;
    z[m * 1024 + ncol] = acc.f[v];
  }
}

// ---------------------------------------------------------------------------
// K2a: LSTM gates -> key_list (h_new), cc (c_new).
// keyH written in B-fragment order for K6 (K-dim = b [4 ksteps], N = u [16 ut]):
//   keyH[ ((ks*16 + ut)*32 + lane)*16 + e ]
// ---------------------------------------------------------------------------
__global__ void k_gates(const float* __restrict__ z, const float* __restrict__ bias,
                        const float* __restrict__ cc_tm1,
                        float* __restrict__ out_key, float* __restrict__ out_cc,
                        _Float16* __restrict__ keyH) {
  int idx = blockIdx.x * blockDim.x + threadIdx.x;   // B*U = 32768
  if (idx >= BB * UU) return;
  int b = idx >> 8, j = idx & 255;
  const float* zr = z + b * 1024;
  float zi = zr[j]       + bias[j];
  float zf = zr[256 + j] + bias[256 + j];
  float zc = zr[512 + j] + bias[512 + j];
  float zo = zr[768 + j] + bias[768 + j];
  float ig = sigmoidf_(zi), fg = sigmoidf_(zf), og = sigmoidf_(zo);
  float cn = fg * cc_tm1[idx] + ig * tanhf(zc);
  float hn = og * tanhf(cn);
  out_key[idx] = hn;
  out_cc[idx]  = cn;
  // fragment-order scatter (K=b, N=j)
  int e, hi;
  invB(b & 31, e, hi);
  int ks = b >> 5, ut = j >> 4, lane = hi * 16 + (j & 15);
  keyH[fragIdx(ks * 16 + ut, lane) + e] = (_Float16)hn;
}

// ---------------------------------------------------------------------------
// K2b: l2-normalized key in B-fragment order for K3 (K-dim = u [8 ksteps], N = b [8 bt]):
//   keyNT[ ((ks*8 + bt)*32 + lane)*16 + e ]
// ---------------------------------------------------------------------------
__global__ void k_keynorm(const float* __restrict__ out_key, _Float16* __restrict__ keyNT) {
  __shared__ float red[256];
  int b = blockIdx.x, j = threadIdx.x;
  float v = out_key[b * 256 + j];
  red[j] = v * v;
  __syncthreads();
  for (int s = 128; s > 0; s >>= 1) {
    if (j < s) red[j] += red[j + s];
    __syncthreads();
  }
  float rinv = rsqrtf(fmaxf(red[0], 1e-12f));
  int e, hi;
  invB(j & 31, e, hi);
  int ks = j >> 5, bt = b >> 4, lane = hi * 16 + (b & 15);
  keyNT[fragIdx(ks * 8 + bt, lane) + e] = (_Float16)(v * rinv);
}

// ---------------------------------------------------------------------------
// K2c: per-row inverse L2 norm of m_tm1 (one wave per row, 8 rows/WG)
// ---------------------------------------------------------------------------
__global__ void k_mnorm(const float* __restrict__ m_tm1, float* __restrict__ inv_norm) {
  int lane = threadIdx.x & 31, wave = threadIdx.x >> 5;
  int row = blockIdx.x * 8 + wave;
  const float* r = m_tm1 + (size_t)row * 256;
  float s = 0.f;
#pragma unroll
  for (int i = 0; i < 8; ++i) { float x = r[lane + i * 32]; s += x * x; }
#pragma unroll
  for (int o = 16; o > 0; o >>= 1) s += __shfl_xor(s, o, 32);
  if (lane == 0) inv_norm[row] = rsqrtf(fmaxf(s, 1e-12f));
}

// ---------------------------------------------------------------------------
// K3: main M-streaming pass, 64 slots per WG (grid 1024, block 128 = 4 waves)
//   cos (WMMA) -> softmax over batch -> c_wr, c_wu (c_ww recomputed inline)
//   + per-column min/argmin partials.
//   m chunk staged in A-fragment order: fragment id = mtile*8 + ks.
// ---------------------------------------------------------------------------
__global__ void __launch_bounds__(128) k_addressing(
    const float* __restrict__ m_tm1, const _Float16* __restrict__ keyNT,
    const float* __restrict__ inv_norm,
    const float* __restrict__ c_wu_tm1, const float* __restrict__ c_wlu_tm1,
    const float* __restrict__ c_wr_tm1, const float* __restrict__ write_gate,
    float* __restrict__ out_cwr, float* __restrict__ out_cwu,
    float* __restrict__ pmin, int* __restrict__ pidx) {
  __shared__ _Float16 mchA[64 * 256];  // 32 KB, A-fragment order (4 mtiles x 8 ks)
  __shared__ float    cosb[64 * 128];  // 32 KB: cos -> c_wr
  int t = threadIdx.x, lane = t & 31, wave = t >> 5, hi = lane >> 4;
  int mbase = blockIdx.x * 64;

  // stage m chunk: coalesced f32 reads, fragment-order f16 scatter into LDS
  const float* msrc = m_tm1 + (size_t)mbase * 256;
  for (int i = 0; i < 128; ++i) {
    int idx = t + i * 128;                 // linear over [64][256]
    int mrow = idx >> 8, k = idx & 255;
    int e, h2; invA(k & 31, e, h2);
    int lane2 = h2 * 16 + (mrow & 15);
    mchA[fragIdx((mrow >> 4) * 8 + (k >> 5), lane2) + e] = (_Float16)msrc[idx];
  }
  __syncthreads();

  // cosine GEMM: wave handles 16 m-rows x all 8 b-tiles, K=256 (8 steps)
  Vf acc[8];
#pragma unroll
  for (int bt = 0; bt < 8; ++bt)
#pragma unroll
    for (int j = 0; j < 8; ++j) acc[bt].f[j] = 0.f;

#pragma unroll
  for (int ks = 0; ks < 8; ++ks) {
    Vh a;
    a.v = *(const v16h*)&mchA[fragIdx(wave * 8 + ks, lane)];
#pragma unroll
    for (int bt = 0; bt < 8; ++bt) {
      Vh bf;
      bf.v = *(const v16h*)&keyNT[fragIdx(ks * 8 + bt, lane)];
      acc[bt].v = wmma32f16(a.v, bf.v, acc[bt].v);
    }
  }
  // scatter C tiles * inv_norm[m] into LDS [64][128]
#pragma unroll
  for (int bt = 0; bt < 8; ++bt)
#pragma unroll
    for (int v = 0; v < 8; ++v) {
      int m = wave * 16 + v + 8 * hi;
      int bcol = bt * 16 + (lane & 15);
      cosb[m * 128 + bcol] = acc[bt].f[v] * inv_norm[mbase + m];
    }
  __syncthreads();

  // softmax over batch (row length 128), write c_wr
  if (t < 64) {
    float* row = cosb + t * 128;
    float mx = row[0];
    for (int i = 1; i < 128; ++i) mx = fmaxf(mx, row[i]);
    float s = 0.f;
    for (int i = 0; i < 128; ++i) { float e = expf(row[i] - mx); row[i] = e; s += e; }
    float rs = 1.f / s;
    float* gout = out_cwr + (size_t)(mbase + t) * 128;
    for (int i = 0; i < 128; ++i) { float v = row[i] * rs; row[i] = v; gout[i] = v; }
  }
  __syncthreads();

  // fused usage update + per-column (batch) min/argmin, thread t owns column t
  float wgs = sigmoidf_(write_gate[0]);
  float minv = 3.4e38f; int mini = mbase;
  for (int i = 0; i < 64; ++i) {
    int idx = i * 128 + t;
    size_t g = (size_t)mbase * 128 + idx;
    float cwr = cosb[idx];
    float cww = wgs * c_wr_tm1[g] + (1.f - wgs) + c_wlu_tm1[g];
    float cwu = 0.95f * c_wu_tm1[g] + cwr + cww;
    out_cwu[g] = cwu;
    if (cwu < minv) { minv = cwu; mini = mbase + i; }
  }
  pmin[blockIdx.x * 128 + t] = minv;
  pidx[blockIdx.x * 128 + t] = mini;
}

// ---------------------------------------------------------------------------
// K4: reduce 1024 partials -> mins[B]; argmin of mins -> i_star (single WG)
// ---------------------------------------------------------------------------
__global__ void __launch_bounds__(128) k_minreduce(
    const float* __restrict__ pmin, const int* __restrict__ pidx,
    float* __restrict__ minsArr, int* __restrict__ istar) {
  __shared__ float sv[128];
  __shared__ int   si[128];
  int t = threadIdx.x;
  float mv = 3.4e38f; int mi = 0;
  for (int p = 0; p < 1024; ++p) {
    float v = pmin[p * 128 + t];
    if (v < mv) { mv = v; mi = pidx[p * 128 + t]; }   // ascending m: first-min
  }
  minsArr[t] = mv; sv[t] = mv; si[t] = mi;
  __syncthreads();
  if (t == 0) {
    float bm = sv[0]; int bi = si[0];
    for (int b = 1; b < 128; ++b)
      if (sv[b] < bm) { bm = sv[b]; bi = si[b]; }
    istar[0] = bi;
  }
}

// ---------------------------------------------------------------------------
// K5: c_wlu = (c_wu <= mins[b]) elementwise
// ---------------------------------------------------------------------------
__global__ void k_wlu(const float* __restrict__ out_cwu, const float* __restrict__ minsArr,
                      float* __restrict__ out_cwlu) {
  int idx = blockIdx.x * 256 + threadIdx.x;   // M*B = 8388608
  float v = out_cwu[idx];
  out_cwlu[idx] = (v <= minsArr[idx & 127]) ? 1.f : 0.f;
}

// ---------------------------------------------------------------------------
// K6: memory = c_ww @ key_list + (m==i_star ? 0 : B) * m_tm1
//   64 m-rows per WG (grid 1024); wave: 16 rows x full U (16 C tiles), K=128.
//   c_ww staged in A-fragment order (4 mtiles x 4 ks); keyH is pre-swizzled.
// ---------------------------------------------------------------------------
__global__ void __launch_bounds__(128) k_memory(
    const float* __restrict__ c_wr_tm1, const float* __restrict__ c_wlu_tm1,
    const float* __restrict__ write_gate, const _Float16* __restrict__ keyH,
    const float* __restrict__ m_tm1, const int* __restrict__ istar,
    float* __restrict__ out_mem) {
  __shared__ _Float16 cww[64 * 128];   // 16 KB, A-fragment order
  int t = threadIdx.x, lane = t & 31, wave = t >> 5, hi = lane >> 4;
  int mbase = blockIdx.x * 64;
  float wgs = sigmoidf_(write_gate[0]);
  for (int i = 0; i < 64; ++i) {
    int idx = t + i * 128;                 // linear over [64][128]
    size_t g = (size_t)mbase * 128 + idx;
    int mrow = idx >> 7, b = idx & 127;
    int e, h2; invA(b & 31, e, h2);
    int lane2 = h2 * 16 + (mrow & 15);
    cww[fragIdx((mrow >> 4) * 4 + (b >> 5), lane2) + e] =
        (_Float16)(wgs * c_wr_tm1[g] + (1.f - wgs) + c_wlu_tm1[g]);
  }
  __syncthreads();

  Vf acc[16];
#pragma unroll
  for (int ut = 0; ut < 16; ++ut)
#pragma unroll
    for (int j = 0; j < 8; ++j) acc[ut].f[j] = 0.f;

#pragma unroll
  for (int ks = 0; ks < 4; ++ks) {
    Vh a;
    a.v = *(const v16h*)&cww[fragIdx(wave * 4 + ks, lane)];
#pragma unroll
    for (int ut = 0; ut < 16; ++ut) {
      Vh bf;
      bf.v = *(const v16h*)&keyH[fragIdx(ks * 16 + ut, lane)];
      acc[ut].v = wmma32f16(a.v, bf.v, acc[ut].v);
    }
  }

  int is = istar[0];
#pragma unroll
  for (int ut = 0; ut < 16; ++ut)
#pragma unroll
    for (int v = 0; v < 8; ++v) {
      int m = mbase + wave * 16 + v + 8 * hi;
      int u = ut * 16 + (lane & 15);
      float keep = (m == is) ? 0.f : 128.f;   // B * (1 - one_hot)
      out_mem[(size_t)m * 256 + u] = acc[ut].f[v] + keep * m_tm1[(size_t)m * 256 + u];
    }
}

// ---------------------------------------------------------------------------
// K7: read partials: c_wr^T @ m_tm1 split over M (32 WGs x 2048 slots each)
//   WG = 8 waves; wave holds 16 b-rows x full U (16 C tiles) in registers.
//   Both chunks staged in fragment order (A: 8 btiles; B: 16 utiles).
//   Deterministic: each WG writes its own partial tile; K8 sums in order.
// ---------------------------------------------------------------------------
__global__ void __launch_bounds__(256) k_read(
    const float* __restrict__ out_cwr, const float* __restrict__ m_tm1,
    float* __restrict__ pread) {
  __shared__ _Float16 cwrchA[32 * 128];  // 8 KB, A-fragment order (frag = btile)
  __shared__ _Float16 mchB[32 * 256];    // 16 KB, B-fragment order (frag = utile)
  int t = threadIdx.x, lane = t & 31, wave = t >> 5, hi = lane >> 4;

  Vf acc[16];
#pragma unroll
  for (int ut = 0; ut < 16; ++ut)
#pragma unroll
    for (int j = 0; j < 8; ++j) acc[ut].f[j] = 0.f;

  int mstart = blockIdx.x * 2048;
  for (int kc = 0; kc < 64; ++kc) {
    int mb = mstart + kc * 32;
    const float* csrc = out_cwr + (size_t)mb * 128;
    const float* msrc = m_tm1 + (size_t)mb * 256;
    // prefetch next chunk (gfx1250 global_prefetch path)
    if (kc + 1 < 64) {
      __builtin_prefetch(csrc + 32 * 128 + (t << 2), 0, 1);
      __builtin_prefetch(msrc + 32 * 256 + (t << 3), 0, 1);
    }
    // A staging: c_wr chunk [32 m][128 b] -> A fragments (rows = b, K = m_local)
    for (int i = 0; i < 16; ++i) {
      int idx = t + i * 256;
      int mloc = idx >> 7, b = idx & 127;
      int e, h2; invA(mloc, e, h2);
      int lane2 = h2 * 16 + (b & 15);
      cwrchA[fragIdx(b >> 4, lane2) + e] = (_Float16)csrc[idx];
    }
    // B staging: m chunk [32 m][256 u] -> B fragments (K = m_local, N = u)
    for (int i = 0; i < 32; ++i) {
      int idx = t + i * 256;
      int mloc = idx >> 8, u = idx & 255;
      int e, h2; invB(mloc, e, h2);
      int lane2 = h2 * 16 + (u & 15);
      mchB[fragIdx(u >> 4, lane2) + e] = (_Float16)msrc[idx];
    }
    __syncthreads();

    Vh a;
    a.v = *(const v16h*)&cwrchA[fragIdx(wave, lane)];
#pragma unroll
    for (int ut = 0; ut < 16; ++ut) {
      Vh bf;
      bf.v = *(const v16h*)&mchB[fragIdx(ut, lane)];
      acc[ut].v = wmma32f16(a.v, bf.v, acc[ut].v);
    }
    __syncthreads();
  }

  float* pout = pread + (size_t)blockIdx.x * (BB * UU);
#pragma unroll
  for (int ut = 0; ut < 16; ++ut)
#pragma unroll
    for (int v = 0; v < 8; ++v) {
      int b = wave * 16 + v + 8 * hi;
      int u = ut * 16 + (lane & 15);
      pout[b * 256 + u] = acc[ut].f[v];
    }
}

// K8: deterministic sum of 32 read partials
__global__ void k_read_reduce(const float* __restrict__ pread, float* __restrict__ out_read) {
  int idx = blockIdx.x * 256 + threadIdx.x;   // 32768
  if (idx >= BB * UU) return;
  float s = 0.f;
  for (int p = 0; p < 32; ++p) s += pread[(size_t)p * (BB * UU) + idx];
  out_read[idx] = s;
}

// ---------------------------------------------------------------------------
extern "C" void kernel_launch(void* const* d_in, const int* in_sizes, int n_in,
                              void* d_out, int out_size, void* d_ws, size_t ws_size,
                              hipStream_t stream) {
  const float* inputs     = (const float*)d_in[0];
  const float* r_tm1      = (const float*)d_in[1];
  const float* m_tm1      = (const float*)d_in[2];
  const float* c_wu_tm1   = (const float*)d_in[3];
  const float* c_wlu_tm1  = (const float*)d_in[4];
  const float* c_wr_tm1   = (const float*)d_in[5];
  const float* h_tm1      = (const float*)d_in[6];
  const float* cc_tm1     = (const float*)d_in[7];
  const float* kernelW    = (const float*)d_in[8];
  const float* rec_kernel = (const float*)d_in[9];
  const float* bias       = (const float*)d_in[10];
  const float* write_gate = (const float*)d_in[11];

  // Output layout (tuple order): read, memory, c_wu, c_wlu, c_wr, key_list, cc
  float* out      = (float*)d_out;
  float* out_read = out;
  float* out_mem  = out_read + BB * UU;
  float* out_cwu  = out_mem + (size_t)MM * UU;
  float* out_cwlu = out_cwu + (size_t)MM * BB;
  float* out_cwr  = out_cwlu + (size_t)MM * BB;
  float* out_key  = out_cwr + (size_t)MM * BB;
  float* out_cc   = out_key + BB * UU;

  // Workspace carve-up (~6 MB)
  char* w = (char*)d_ws;
  float*     z        = (float*)w;     w += (size_t)BB * 1024 * sizeof(float);   // 512 KB
  _Float16*  keyNT    = (_Float16*)w;  w += (size_t)UU * BB * sizeof(_Float16);  // 64 KB (frag order)
  _Float16*  keyH     = (_Float16*)w;  w += (size_t)BB * UU * sizeof(_Float16);  // 64 KB (frag order)
  float*     inv_norm = (float*)w;     w += (size_t)MM * sizeof(float);          // 256 KB
  float*     pmin     = (float*)w;     w += (size_t)1024 * 128 * sizeof(float);  // 512 KB
  int*       pidx     = (int*)w;       w += (size_t)1024 * 128 * sizeof(int);    // 512 KB
  float*     minsArr  = (float*)w;     w += 128 * sizeof(float);
  int*       istar    = (int*)w;       w += 4 * sizeof(int);
  w = (char*)(((uintptr_t)w + 255) & ~(uintptr_t)255);
  float*     pread    = (float*)w;     w += (size_t)32 * BB * UU * sizeof(float); // 4 MB

  // 1) LSTM GEMM + gates + key normalization + memory row norms
  k_lstm_gemm<<<128, 128, 0, stream>>>(inputs, r_tm1, h_tm1, kernelW, rec_kernel, z);
  k_gates<<<(BB * UU + 255) / 256, 256, 0, stream>>>(z, bias, cc_tm1, out_key, out_cc, keyH);
  k_keynorm<<<BB, 256, 0, stream>>>(out_key, keyNT);
  k_mnorm<<<MM / 8, 256, 0, stream>>>(m_tm1, inv_norm);

  // 2) Main M-streaming pass: cos -> softmax -> c_wr, c_wu, min partials
  k_addressing<<<MM / 64, 128, 0, stream>>>(m_tm1, keyNT, inv_norm, c_wu_tm1, c_wlu_tm1,
                                            c_wr_tm1, write_gate, out_cwr, out_cwu,
                                            pmin, pidx);

  // 3) Global min / argmin, then c_wlu
  k_minreduce<<<1, 128, 0, stream>>>(pmin, pidx, minsArr, istar);
  k_wlu<<<(MM * BB) / 256, 256, 0, stream>>>(out_cwu, minsArr, out_cwlu);

  // 4) Memory update GEMM (fused keep*m_tm1 add)
  k_memory<<<MM / 64, 128, 0, stream>>>(c_wr_tm1, c_wlu_tm1, write_gate, keyH, m_tm1,
                                        istar, out_mem);

  // 5) read = c_wr^T @ m_tm1 (deterministic split-K) + reduce
  k_read<<<32, 256, 0, stream>>>(out_cwr, m_tm1, pread);
  k_read_reduce<<<(BB * UU + 255) / 256, 256, 0, stream>>>(pread, out_read);
}
